// Kmodel_90537910599714
// MI455X (gfx1250) — matrix operand order repaired and checked by reading, not compiled, not run on hardware
//
#include <hip/hip_runtime.h>
#include <hip/hip_bf16.h>
#include <math.h>

// ---------------- constants ----------------
#define BB   16
#define TT   127
#define EE   256
#define HH   512
#define GG   2048      // 4*H
#define VV   32000
#define MP   2048      // padded B*T (2032 -> 2048)
#define D2H  1024      // 2*H

typedef __attribute__((ext_vector_type(16))) _Float16 v16h;
typedef __attribute__((ext_vector_type(8)))  _Float16 v8h;
typedef __attribute__((ext_vector_type(8)))  float    v8f;

__device__ __forceinline__ float hsig(float x) {
    return fminf(fmaxf(0.2f * x + 0.5f, 0.0f), 1.0f);
}

__device__ __forceinline__ v16h make_frag(const _Float16* lo, const _Float16* hi) {
    v8h a = *(const v8h*)lo;
    v8h b = *(const v8h*)hi;
    v16h r;
#pragma unroll
    for (int i = 0; i < 8; ++i) { r[i] = a[i]; r[8 + i] = b[i]; }
    return r;
}

// ---------------- prep kernels ----------------

// x_h[m][e], m = t*16 + b, rows >= 2032 zeroed
__global__ __launch_bounds__(256) void gather_x_kernel(
    const int* __restrict__ tokens, const float* __restrict__ emb,
    _Float16* __restrict__ xh)
{
    int idx = blockIdx.x * 256 + threadIdx.x;      // MP*EE total
    if (idx >= MP * EE) return;
    int m = idx >> 8;           // /EE
    int e = idx & (EE - 1);
    _Float16 v = (_Float16)0.0f;
    if (m < BB * TT) {
        int t = m >> 4, b = m & 15;
        int tok = tokens[b * TT + t];
        v = (_Float16)emb[(size_t)tok * EE + e];
    }
    xh[idx] = v;
}

// dst[n*K + k] = (f16) src[k*N + n]
__global__ __launch_bounds__(256) void transpose_f32_to_f16(
    const float* __restrict__ src, _Float16* __restrict__ dst, int K, int N)
{
    int idx = blockIdx.x * 256 + threadIdx.x;
    if (idx >= K * N) return;
    int k = idx / N;
    int n = idx - k * N;
    dst[(size_t)n * K + k] = (_Float16)src[idx];
}

// ---------------- generic WMMA GEMM: C[Mp][N] = A[Mp][K](f16) * Bt[N][K](f16) + bias ----------------
// one 16x16 tile per wave; block = 256 threads = 8 waves; grid = (N/128, Mp/16)
__global__ __launch_bounds__(256) void gemm_f16_kernel(
    const _Float16* __restrict__ A, const _Float16* __restrict__ Bt,
    const float* __restrict__ bias, float* __restrict__ C, int K, int N)
{
    int wid  = threadIdx.x >> 5;
    int lane = threadIdx.x & 31;
    int n0 = (blockIdx.x * 8 + wid) * 16;
    int m0 = blockIdx.y * 16;
    int ncol  = lane & 15;
    int abase = (lane < 16) ? 0 : 8;    // A-frag K base
    int bbase = (lane < 16) ? 0 : 8;    // B-frag K base
    int mAdd  = (lane < 16) ? 0 : 8;    // C-frag row offset

    float bv = bias[n0 + ncol];
    v8f acc;
#pragma unroll
    for (int j = 0; j < 8; ++j) acc[j] = bv;

    const _Float16* ap = A  + (size_t)(m0 + ncol) * K;
    const _Float16* bp = Bt + (size_t)(n0 + ncol) * K;
    for (int k0 = 0; k0 < K; k0 += 32) {
        v16h a = make_frag(ap + k0 + abase, ap + k0 + abase + 16);
        v16h b = make_frag(bp + k0 + bbase, bp + k0 + bbase + 16);
        acc = __builtin_amdgcn_wmma_f32_16x16x32_f16(false, a, false, b,
                                                     (short)0, acc, false, false);
    }
#pragma unroll
    for (int j = 0; j < 8; ++j)
        C[(size_t)(m0 + j + mAdd) * N + n0 + ncol] = acc[j];
}

// ---------------- bidirectional LSTM scan ----------------
// grid = 2 blocks (dir 0 = fwd, dir 1 = bwd), 1024 threads = 32 waves.
// wave w owns H-columns [w*16, w*16+16) of ALL FOUR gates -> gate math and the
// cell state c stay in registers as aligned WMMA C-fragments. h lives in LDS.
__global__ __launch_bounds__(1024) void lstm_scan_kernel(
    const float* __restrict__ pre_f, const float* __restrict__ pre_b,
    const _Float16* __restrict__ rf, const _Float16* __restrict__ rb,
    _Float16* __restrict__ h_cat)
{
    const int dir = blockIdx.x;
    const float*    pre = dir ? pre_b : pre_f;   // [MP][GG], row = t*16+b
    const _Float16* Rt  = dir ? rb : rf;         // [GG][HH] (transposed)
    const int colOff = dir ? HH : 0;

    __shared__ _Float16 hbuf[16 * HH];           // 16 KB

    const int tid  = threadIdx.x;
    const int wid  = tid >> 5;
    const int lane = tid & 31;
    const int ncol  = lane & 15;
    const int abase = (lane < 16) ? 0 : 8;
    const int bbase = (lane < 16) ? 0 : 8;
    const int mAdd  = (lane < 16) ? 0 : 8;
    const int hc    = wid * 16 + ncol;           // column in H (0..511)
    const int arow  = lane & 15;                 // A-frag row (M)

    for (int i = tid; i < 16 * HH; i += 1024) hbuf[i] = (_Float16)0.0f;
    v8f c = {};
    __syncthreads();

    for (int step = 0; step < TT; ++step) {
        const int t = dir ? (TT - 1 - step) : step;
        const float* prow = pre + (size_t)t * 16 * GG;

        v8f acc[4];
#pragma unroll
        for (int g = 0; g < 4; ++g)
#pragma unroll
            for (int j = 0; j < 8; ++j)
                acc[g][j] = prow[(size_t)(j + mAdd) * GG + g * HH + hc];

        for (int kk = 0; kk < 16; ++kk) {        // K = 512 in steps of 32
            const int k0 = kk * 32;
            v16h a = make_frag(&hbuf[arow * HH + k0 + abase],
                               &hbuf[arow * HH + k0 + abase + 16]);
#pragma unroll
            for (int g = 0; g < 4; ++g) {
                const _Float16* bp = Rt + (size_t)(g * HH + hc) * HH + k0 + bbase;
                v16h b = make_frag(bp, bp + 16);
                acc[g] = __builtin_amdgcn_wmma_f32_16x16x32_f16(
                    false, a, false, b, (short)0, acc[g], false, false);
            }
        }
        __syncthreads();                         // all hbuf reads done

#pragma unroll
        for (int j = 0; j < 8; ++j) {
            float iv = hsig(acc[0][j]);
            float fv = hsig(acc[1][j]);
            float gv = tanhf(acc[2][j]);
            float ov = hsig(acc[3][j]);
            float cv = fv * c[j] + iv * gv;
            c[j] = cv;
            float hv = ov * tanhf(cv);
            int row = j + mAdd;
            _Float16 hh = (_Float16)hv;
            hbuf[row * HH + hc] = hh;
            h_cat[(size_t)(t * 16 + row) * D2H + colOff + hc] = hh;
        }
        __syncthreads();                         // new h visible
    }
}

// ---------------- output dense: logits = h_cat * Wd_t + bd, remapped rows ----------------
// Register-blocked: each wave computes 4 M-tiles x 2 N-tiles (8 accumulators).
// Block = 256 threads = 8 waves -> 64 rows x 256 cols per block.
// grid = (V/256 = 125, MP/64 = 32); padded M-tiles (t >= 127) are computed on
// garbage but never stored.
#define DMT 4
#define DNT 2
__global__ __launch_bounds__(256) void dense_kernel(
    const _Float16* __restrict__ A, const _Float16* __restrict__ Bt,
    const float* __restrict__ bd, float* __restrict__ out)
{
    int wid  = threadIdx.x >> 5;
    int lane = threadIdx.x & 31;
    int ncol  = lane & 15;
    int abase = (lane < 16) ? 0 : 8;
    int bbase = (lane < 16) ? 0 : 8;
    int mAdd  = (lane < 16) ? 0 : 8;

    int mBase = blockIdx.y * (DMT * 16);            // 64-row block
    int nBase = blockIdx.x * 256 + wid * (DNT * 16);// 32 cols per wave

    v8f acc[DMT][DNT];
#pragma unroll
    for (int nt = 0; nt < DNT; ++nt) {
        float bv = bd[nBase + nt * 16 + ncol];
#pragma unroll
        for (int mt = 0; mt < DMT; ++mt)
#pragma unroll
            for (int j = 0; j < 8; ++j) acc[mt][nt][j] = bv;
    }

    const _Float16* ap[DMT];
#pragma unroll
    for (int mt = 0; mt < DMT; ++mt)
        ap[mt] = A + (size_t)(mBase + mt * 16 + ncol) * D2H;
    const _Float16* bp[DNT];
#pragma unroll
    for (int nt = 0; nt < DNT; ++nt)
        bp[nt] = Bt + (size_t)(nBase + nt * 16 + ncol) * D2H;

    for (int k0 = 0; k0 < D2H; k0 += 32) {
        v16h a[DMT], b[DNT];
#pragma unroll
        for (int mt = 0; mt < DMT; ++mt)
            a[mt] = make_frag(ap[mt] + k0 + abase, ap[mt] + k0 + abase + 16);
#pragma unroll
        for (int nt = 0; nt < DNT; ++nt)
            b[nt] = make_frag(bp[nt] + k0 + bbase, bp[nt] + k0 + bbase + 16);
#pragma unroll
        for (int mt = 0; mt < DMT; ++mt)
#pragma unroll
            for (int nt = 0; nt < DNT; ++nt)
                acc[mt][nt] = __builtin_amdgcn_wmma_f32_16x16x32_f16(
                    false, a[mt], false, b[nt], (short)0, acc[mt][nt], false, false);
    }

#pragma unroll
    for (int mt = 0; mt < DMT; ++mt) {
        int m0 = mBase + mt * 16;
        int t  = m0 >> 4;                 // all rows of a tile share t
        if (t >= TT) continue;            // padded tiles: skip store
#pragma unroll
        for (int nt = 0; nt < DNT; ++nt)
#pragma unroll
            for (int j = 0; j < 8; ++j) {
                int b = j + mAdd;         // batch index within tile
                out[(size_t)(b * TT + t) * VV + nBase + nt * 16 + ncol] = acc[mt][nt][j];
            }
    }
}

// ---------------- in-place row softmax over d_out ----------------
__global__ __launch_bounds__(256) void softmax_kernel(float* __restrict__ out)
{
    float* p = out + (size_t)blockIdx.x * VV;
    __shared__ float red[256];
    int tid = threadIdx.x;

    float m = -INFINITY;
    for (int i = tid; i < VV; i += 256) m = fmaxf(m, p[i]);
    red[tid] = m; __syncthreads();
    for (int s = 128; s > 0; s >>= 1) {
        if (tid < s) red[tid] = fmaxf(red[tid], red[tid + s]);
        __syncthreads();
    }
    m = red[0]; __syncthreads();

    float sum = 0.0f;
    for (int i = tid; i < VV; i += 256) sum += __expf(p[i] - m);
    red[tid] = sum; __syncthreads();
    for (int s = 128; s > 0; s >>= 1) {
        if (tid < s) red[tid] += red[tid + s];
        __syncthreads();
    }
    float inv = 1.0f / red[0];

    for (int i = tid; i < VV; i += 256) p[i] = __expf(p[i] - m) * inv;
}

// ---------------- launcher ----------------
extern "C" void kernel_launch(void* const* d_in, const int* in_sizes, int n_in,
                              void* d_out, int out_size, void* d_ws, size_t ws_size,
                              hipStream_t stream)
{
    (void)in_sizes; (void)n_in; (void)out_size; (void)ws_size;
    const int*   tokens = (const int*)  d_in[0];
    const float* emb    = (const float*)d_in[1];
    const float* k_fwd  = (const float*)d_in[2];
    const float* r_fwd  = (const float*)d_in[3];
    const float* b_fwd  = (const float*)d_in[4];
    const float* k_bwd  = (const float*)d_in[5];
    const float* r_bwd  = (const float*)d_in[6];
    const float* b_bwd  = (const float*)d_in[7];
    const float* Wd     = (const float*)d_in[8];
    const float* bd     = (const float*)d_in[9];
    float* out = (float*)d_out;

    // workspace layout (bytes)
    char* ws = (char*)d_ws;
    size_t o = 0;
    _Float16* x_h  = (_Float16*)(ws + o); o += (size_t)MP * EE  * 2;   // 1 MB
    _Float16* kf_t = (_Float16*)(ws + o); o += (size_t)GG * EE  * 2;   // 1 MB
    _Float16* kb_t = (_Float16*)(ws + o); o += (size_t)GG * EE  * 2;   // 1 MB
    _Float16* rf_t = (_Float16*)(ws + o); o += (size_t)GG * HH  * 2;   // 2 MB
    _Float16* rb_t = (_Float16*)(ws + o); o += (size_t)GG * HH  * 2;   // 2 MB
    _Float16* hcat = (_Float16*)(ws + o); o += (size_t)MP * D2H * 2;   // 4 MB
    float*    pref = (float*)   (ws + o); o += (size_t)MP * GG  * 4;   // 16 MB
    float*    preb = (float*)   (ws + o); o += (size_t)MP * GG  * 4;   // 16 MB
    _Float16* wd_t = (_Float16*)(ws + o); o += (size_t)VV * D2H * 2;   // 62.5 MB

    // 1) prep: gather + transposed f16 weight copies
    gather_x_kernel<<<(MP * EE) / 256, 256, 0, stream>>>(tokens, emb, x_h);
    transpose_f32_to_f16<<<(EE  * GG) / 256, 256, 0, stream>>>(k_fwd, kf_t, EE,  GG);
    transpose_f32_to_f16<<<(EE  * GG) / 256, 256, 0, stream>>>(k_bwd, kb_t, EE,  GG);
    transpose_f32_to_f16<<<(HH  * GG) / 256, 256, 0, stream>>>(r_fwd, rf_t, HH,  GG);
    transpose_f32_to_f16<<<(HH  * GG) / 256, 256, 0, stream>>>(r_bwd, rb_t, HH,  GG);
    transpose_f32_to_f16<<<(D2H * VV) / 256, 256, 0, stream>>>(Wd,    wd_t, D2H, VV);

    // 2) input projections: pre = x_h @ K^T + b   (M=2048, N=2048, K=256)
    gemm_f16_kernel<<<dim3(GG / 128, MP / 16), 256, 0, stream>>>(x_h, kf_t, b_fwd, pref, EE, GG);
    gemm_f16_kernel<<<dim3(GG / 128, MP / 16), 256, 0, stream>>>(x_h, kb_t, b_bwd, preb, EE, GG);

    // 3) bidirectional recurrent scan (2 persistent workgroups)
    lstm_scan_kernel<<<2, 1024, 0, stream>>>(pref, preb, rf_t, rb_t, hcat);

    // 4) output dense -> logits straight into d_out (row remap t*16+b -> b*T+t)
    dense_kernel<<<dim3(VV / 256, MP / 64), 256, 0, stream>>>(hcat, wd_t, bd, out);

    // 5) in-place softmax per (b,t) row
    softmax_kernel<<<BB * TT, 256, 0, stream>>>(out);
}
